// MultiHeadAttention_2284922602069
// MI455X (gfx1250) — compile-verified, matrix operand order checked by reading, not run on hardware
//
#include <hip/hip_runtime.h>

// ---------------------------------------------------------------------------
// MHA for gfx1250 (MI455X): bf16 WMMA everywhere, flash-softmax attention with
// workgroup-cooperative K/V staging in LDS (async-to-LDS DMA path).
// Shapes fixed by the reference: B=2, S=2048, D_MODEL=1024, H=16, D_K=64.
// ---------------------------------------------------------------------------

typedef __attribute__((ext_vector_type(16))) __bf16     v16bf;
typedef __attribute__((ext_vector_type(8)))  float      v8f;
typedef __attribute__((ext_vector_type(4)))  unsigned   v4u;
typedef __attribute__((ext_vector_type(4)))  int        v4i;

#define BATCH   2
#define SEQ     2048
#define DMODEL  1024
#define NHEAD   16
#define DK      64

// ---- bf16 helpers (RNE, no reliance on __bf16 arithmetic) -----------------
static __device__ __forceinline__ __bf16 f2bf(float f) {
  unsigned u = __builtin_bit_cast(unsigned, f);
  unsigned r = u + 0x7FFFu + ((u >> 16) & 1u);
  unsigned short h = (unsigned short)(r >> 16);
  return __builtin_bit_cast(__bf16, h);
}

union FragU { v4u u[2]; v16bf v; };

// Load a 16x32 bf16 WMMA operand fragment from a row-major matrix (global or LDS).
// ISA 16-bit A/B layout: lane<16 -> K = {0..7, 16..23}; lane>=16 -> K = {8..15, 24..31}.
// Caller passes row = tile_row_base + (lane & 15).
static __device__ __forceinline__ v16bf frag_rm(const __bf16* base, int row,
                                                int ld, int k0, int lane) {
  const __bf16* p = base + (size_t)row * ld + k0 + ((lane >> 4) << 3);
  FragU r;
  r.u[0] = *(const v4u*)(p);
  r.u[1] = *(const v4u*)(p + 16);
  return r.v;
}

static __device__ __forceinline__ v8f wmma_bf16(v16bf a, v16bf b, v8f c) {
  return __builtin_amdgcn_wmma_f32_16x16x32_bf16(false, a, false, b,
                                                 (short)0, c, false, false);
}

// ---- 16-byte global -> LDS copy: async DMA path when the toolchain has it --
#if __has_builtin(__builtin_amdgcn_global_load_async_to_lds_b128)
#define MHA_ASYNC_LDS 1
#else
#define MHA_ASYNC_LDS 0
#endif

static __device__ __forceinline__ void cp16_g2l(const __bf16* g, __bf16* l) {
#if MHA_ASYNC_LDS
  // Signature (probe-confirmed): (int4* global_src, int4* lds_dst, imm, cpol)
  __builtin_amdgcn_global_load_async_to_lds_b128((v4i*)g, (v4i*)l, 0, 0);
#else
  *(v4u*)l = *(const v4u*)g;
#endif
}

static __device__ __forceinline__ void wait_async_lds() {
#if MHA_ASYNC_LDS
#if __has_builtin(__builtin_amdgcn_s_wait_asynccnt)
  __builtin_amdgcn_s_wait_asynccnt(0);
#else
  asm volatile("s_wait_asynccnt 0" ::: "memory");
#endif
#endif
}

// ---------------------------------------------------------------------------
// fp32 -> bf16 convert (activations + weights)
// ---------------------------------------------------------------------------
__global__ void mha_cvt_bf16(const float* __restrict__ in,
                             __bf16* __restrict__ out, int n) {
  for (int i = blockIdx.x * blockDim.x + threadIdx.x; i < n;
       i += gridDim.x * blockDim.x)
    out[i] = f2bf(in[i]);
}

// ---------------------------------------------------------------------------
// Y = X * W^T + bias.  X:[M,K] bf16 row-major. W:[N,K] bf16 row-major
// (row n of W == column n of W^T, so B-fragments load contiguously).
// One wave per 16x16 output tile; K-loop of v_wmma_f32_16x16x32_bf16.
// mode 0: out bf16 as [B,H,S,DK]      (Q, K projections)
// mode 1: out bf16 as [B,H,DK,S]      (V projection -> V^T per head)
// mode 2: out fp32 as [M,N]           (final O projection)
// ---------------------------------------------------------------------------
__global__ void mha_gemm_wmma(const __bf16* __restrict__ A,
                              const __bf16* __restrict__ W,
                              const float* __restrict__ bias,
                              void* __restrict__ out,
                              int M, int N, int K, int mode) {
  const int lane = threadIdx.x & 31;
  const int l15  = lane & 15;
  const int half = lane >> 4;

  const int wid = blockIdx.x * (blockDim.x >> 5) + (threadIdx.x >> 5);
  const int ntn = N >> 4;
  const int nt  = wid % ntn;
  const int mt  = wid / ntn;
  if (mt >= (M >> 4)) return;

  const int m0 = mt << 4, n0 = nt << 4;
  const __bf16* arow = A + (size_t)(m0 + l15) * K;
  const __bf16* brow = W + (size_t)(n0 + l15) * K;

  v8f c = {};
  for (int k0 = 0; k0 < K; k0 += 32) {
    __builtin_prefetch(arow + k0 + 256, 0, 3);   // global_prefetch_b8 (near)
    __builtin_prefetch(brow + k0 + 256, 0, 3);
    v16bf a = frag_rm(A, m0 + l15, K, k0, lane);
    v16bf b = frag_rm(W, n0 + l15, K, k0, lane);
    c = wmma_bf16(a, b, c);
  }

  const int n = n0 + l15;                 // C/D layout: col = lane % 16
  const float bn = bias[n];
  if (mode == 2) {
    float* o = (float*)out;
#pragma unroll
    for (int i = 0; i < 8; ++i) {
      int m = m0 + (half << 3) + i;       // C/D layout: row = (lane/16)*8 + i
      o[(size_t)m * N + n] = c[i] + bn;
    }
  } else {
    __bf16* o = (__bf16*)out;
    const int h = n >> 6, d = n & 63;
#pragma unroll
    for (int i = 0; i < 8; ++i) {
      int m = m0 + (half << 3) + i;
      int b_ = m >> 11, s = m & (SEQ - 1);
      size_t idx = (mode == 0)
        ? (((size_t)(b_ * NHEAD + h) * SEQ + s) * DK + d)        // [B,H,S,DK]
        : (((size_t)(b_ * NHEAD + h) * DK + d) * SEQ + s);       // [B,H,DK,S]
      o[idx] = f2bf(c[i] + bn);
    }
  }
}

// ---------------------------------------------------------------------------
// Flash attention core. One 128-thread workgroup (4 waves) owns one (b,h) and
// 64 q-rows; each wave does a 16-row q tile. Per 32-column KV slab:
//   - cooperative stage of K slab (32x64) and V^T slab (64x32) into LDS
//     (async-to-LDS), shared by all 4 waves (4x L2 traffic cut)
//   - scores via WMMA from LDS K, online softmax, P slab through LDS to
//     reshape C-layout -> A-fragment, context WMMA against LDS V^T.
// ---------------------------------------------------------------------------
__global__ void mha_attn_wmma(const __bf16* __restrict__ Q,
                              const __bf16* __restrict__ Km,
                              const __bf16* __restrict__ Vt,
                              const int* __restrict__ mask,
                              __bf16* __restrict__ ctx) {
  __shared__ __align__(64) __bf16 kbuf[32 * 64];       // K slab   (4 KB)
  __shared__ __align__(64) __bf16 vbuf[64 * 32];       // V^T slab (4 KB)
  __shared__ __align__(64) __bf16 pbuf[4 * 16 * 32];   // P slabs  (4 KB)

  const int tid  = threadIdx.x;
  const int lane = tid & 31;
  const int l15  = lane & 15;
  const int half = lane >> 4;
  const int wv   = tid >> 5;                 // wave id in block (0..3)

  const int nqb = SEQ / 64;                  // 64 q-rows per block
  const int bh  = blockIdx.x / nqb;
  const int qb  = blockIdx.x % nqb;
  const int b_  = bh / NHEAD, h = bh % NHEAD;
  const int q0  = qb * 64 + wv * 16;

  const __bf16* Qh  = Q  + (size_t)bh * SEQ * DK;
  const __bf16* Kh  = Km + (size_t)bh * SEQ * DK;
  const __bf16* Vth = Vt + (size_t)bh * DK * SEQ;
  const int*    mb  = mask + (size_t)b_ * SEQ * SEQ;
  __bf16* pl = pbuf + wv * (16 * 32);

  // Q fragments are loop-invariant (16 rows x 64 d)
  const v16bf aq0 = frag_rm(Qh, q0 + l15, DK, 0,  lane);
  const v16bf aq1 = frag_rm(Qh, q0 + l15, DK, 32, lane);

  float rmax[8], rsum[8];
  v8f acc[4] = {{}, {}, {}, {}};
#pragma unroll
  for (int i = 0; i < 8; ++i) { rmax[i] = -3.0e38f; rsum[i] = 0.f; }

  for (int t = 0; t < SEQ / 32; ++t) {
    const int kt = t * 32;

    // ---- cooperative slab staging: 256 chunks of 16B each buffer ----------
    {
      const __bf16* gk = Kh + (size_t)kt * DK;   // K slab is fully contiguous
#pragma unroll
      for (int j = 0; j < 2; ++j) {
        const int c = tid + j * 128;             // chunk 0..255
        cp16_g2l(gk + c * 8, kbuf + c * 8);
        const int r = c >> 2, co = (c & 3) * 8;  // V^T: 4 chunks per d-row
        cp16_g2l(Vth + (size_t)r * SEQ + kt + co, vbuf + r * 32 + co);
      }
      wait_async_lds();
      __syncthreads();
    }

    // ---- scores: two 16x16 tiles from LDS K slab --------------------------
    v8f s0 = {}, s1 = {};
    s0 = wmma_bf16(aq0, frag_rm(kbuf, l15,      DK, 0,  lane), s0);
    s0 = wmma_bf16(aq1, frag_rm(kbuf, l15,      DK, 32, lane), s0);
    s1 = wmma_bf16(aq0, frag_rm(kbuf, 16 + l15, DK, 0,  lane), s1);
    s1 = wmma_bf16(aq1, frag_rm(kbuf, 16 + l15, DK, 32, lane), s1);

    // ---- scale (1/sqrt(64)), mask, online softmax -------------------------
#pragma unroll
    for (int i = 0; i < 8; ++i) {
      const int rq = q0 + (half << 3) + i;
      float v0 = s0[i] * 0.125f;
      float v1 = s1[i] * 0.125f;
      if (mb[(size_t)rq * SEQ + kt + l15] == 0)      v0 = -1.0e9f;
      if (mb[(size_t)rq * SEQ + kt + 16 + l15] == 0) v1 = -1.0e9f;

      float mx = fmaxf(v0, v1);
      mx = fmaxf(mx, __shfl_xor(mx, 1, 32));
      mx = fmaxf(mx, __shfl_xor(mx, 2, 32));
      mx = fmaxf(mx, __shfl_xor(mx, 4, 32));
      mx = fmaxf(mx, __shfl_xor(mx, 8, 32));   // reduce within 16-lane group

      const float mnew = fmaxf(rmax[i], mx);
      const float corr = __expf(rmax[i] - mnew);
      const float p0 = __expf(v0 - mnew);
      const float p1 = __expf(v1 - mnew);
      float ps = p0 + p1;
      ps += __shfl_xor(ps, 1, 32);
      ps += __shfl_xor(ps, 2, 32);
      ps += __shfl_xor(ps, 4, 32);
      ps += __shfl_xor(ps, 8, 32);

      rsum[i] = rsum[i] * corr + ps;
      rmax[i] = mnew;
#pragma unroll
      for (int n = 0; n < 4; ++n) acc[n][i] *= corr;

      const int m = (half << 3) + i;           // C-layout row
      pl[m * 32 + l15]      = f2bf(p0);        // P slab: 16 rows x 32 cols
      pl[m * 32 + 16 + l15] = f2bf(p1);
    }

    asm volatile("s_wait_dscnt 0" ::: "memory");   // P slab visible in LDS

    // ---- context: P (A-frag from LDS) x V^T slab (B-frags from LDS) -------
    {
      const __bf16* pp = pl + l15 * 32 + (half << 3);
      FragU r;
      r.u[0] = *(const v4u*)(pp);
      r.u[1] = *(const v4u*)(pp + 16);
      const v16bf ap = r.v;
#pragma unroll
      for (int n = 0; n < 4; ++n) {
        v16bf bv = frag_rm(vbuf, n * 16 + l15, 32, 0, lane);
        acc[n] = wmma_bf16(ap, bv, acc[n]);
      }
    }
    __syncthreads();   // all waves done with slabs before restaging
  }

  // normalize and scatter context back to [B,S,H*DK] (bf16 for final GEMM)
#pragma unroll
  for (int i = 0; i < 8; ++i) {
    const float inv = 1.0f / rsum[i];
    const int   s   = q0 + (half << 3) + i;
#pragma unroll
    for (int n = 0; n < 4; ++n) {
      const int d = n * 16 + l15;
      ctx[((size_t)(b_ * SEQ + s)) * DMODEL + h * DK + d] = f2bf(acc[n][i] * inv);
    }
  }
}

// ---------------------------------------------------------------------------
// Host launcher
// ---------------------------------------------------------------------------
extern "C" void kernel_launch(void* const* d_in, const int* in_sizes, int n_in,
                              void* d_out, int out_size, void* d_ws, size_t ws_size,
                              hipStream_t stream) {
  (void)in_sizes; (void)n_in; (void)out_size; (void)ws_size;

  const int M  = BATCH * SEQ;        // 4096
  const int NA = M * DMODEL;         // 4,194,304 activation elements
  const int NW = DMODEL * DMODEL;    // 1,048,576 weight elements

  __bf16* ws = (__bf16*)d_ws;
  __bf16* Xq  = ws;                 // bf16 activations
  __bf16* Xk  = Xq  + NA;
  __bf16* Xv  = Xk  + NA;
  __bf16* Wq  = Xv  + NA;           // bf16 weights
  __bf16* Wk  = Wq  + NW;
  __bf16* Wv  = Wk  + NW;
  __bf16* Wo  = Wv  + NW;
  __bf16* Qb  = Wo  + NW;           // [B,H,S,DK]
  __bf16* Kb  = Qb  + NA;           // [B,H,S,DK]
  __bf16* Vtb = Kb  + NA;           // [B,H,DK,S]
  __bf16* Ctx = Vtb + NA;           // [B,S,DMODEL]

  const float* q  = (const float*)d_in[0];
  const float* k  = (const float*)d_in[1];
  const float* v  = (const float*)d_in[2];
  const int*   mk = (const int*)  d_in[3];
  const float* wq = (const float*)d_in[4];  const float* bq = (const float*)d_in[5];
  const float* wk = (const float*)d_in[6];  const float* bk = (const float*)d_in[7];
  const float* wv = (const float*)d_in[8];  const float* bv = (const float*)d_in[9];
  const float* wo = (const float*)d_in[10]; const float* bo = (const float*)d_in[11];

  // 1) fp32 -> bf16
  mha_cvt_bf16<<<4096, 256, 0, stream>>>(q,  Xq, NA);
  mha_cvt_bf16<<<4096, 256, 0, stream>>>(k,  Xk, NA);
  mha_cvt_bf16<<<4096, 256, 0, stream>>>(v,  Xv, NA);
  mha_cvt_bf16<<<1024, 256, 0, stream>>>(wq, Wq, NW);
  mha_cvt_bf16<<<1024, 256, 0, stream>>>(wk, Wk, NW);
  mha_cvt_bf16<<<1024, 256, 0, stream>>>(wv, Wv, NW);
  mha_cvt_bf16<<<1024, 256, 0, stream>>>(wo, Wo, NW);

  // 2) Q/K/V projections (WMMA).  waves = (M/16)*(DMODEL/16) = 16384; 4/block
  const int gemm_blocks = (M / 16) * (DMODEL / 16) / 4;
  mha_gemm_wmma<<<gemm_blocks, 128, 0, stream>>>(Xq, Wq, bq, Qb,  M, DMODEL, DMODEL, 0);
  mha_gemm_wmma<<<gemm_blocks, 128, 0, stream>>>(Xk, Wk, bk, Kb,  M, DMODEL, DMODEL, 0);
  mha_gemm_wmma<<<gemm_blocks, 128, 0, stream>>>(Xv, Wv, bv, Vtb, M, DMODEL, DMODEL, 1);

  // 3) flash attention: 1 block per (b, h, 64 q-rows) = 2*16*32 = 1024 blocks
  mha_attn_wmma<<<BATCH * NHEAD * (SEQ / 64), 128, 0, stream>>>(Qb, Kb, Vtb, mk, Ctx);

  // 4) output projection -> fp32 d_out
  mha_gemm_wmma<<<gemm_blocks, 128, 0, stream>>>(Ctx, Wo, bo, d_out, M, DMODEL, DMODEL, 2);
}
